// VideoMamba_46033459479202
// MI455X (gfx1250) — matrix operand order repaired and checked by reading, not compiled, not run on hardware
//
#include <hip/hip_runtime.h>
#include <hip/hip_bf16.h>
#include <math.h>

#define D_MODEL 192
#define D_STATE 16
#define D_CONV 4
#define D_INNER 384
#define DT_RANK 12
#define DEPTH 8
#define SEQ_T 8
#define GH 14
#define GW 14
#define NPATCH (SEQ_T*GH*GW)         // 1568
#define LTOK   (1 + NPATCH)          // 1569
#define XDBL_N (DT_RANK + 2*D_STATE) // 44
#define IMGSZ 224
#define KPAD 32                      // padded K for the delta GEMM
#define GEMM_NT 4                    // 16x64 output strip per wave

typedef __attribute__((ext_vector_type(16))) __bf16 v16bf;
typedef __attribute__((ext_vector_type(8)))  __bf16 v8bf;   // 16 bytes -> b128 load
typedef __attribute__((ext_vector_type(8)))  float  v8f;

union frag16 { v16bf v; v8bf h[2]; };

__device__ __forceinline__ float sigmoidf_(float x){ return 1.0f/(1.0f+__expf(-x)); }
__device__ __forceinline__ float siluf_(float x){ return x*sigmoidf_(x); }

// load one K-step's fragment set: A (2 x b128) + GEMM_NT B fragments (2 x b128 each)
__device__ __forceinline__ void load_frags(const __bf16* __restrict__ arow,
                                           const __bf16* __restrict__ const* wrow,
                                           int k0, int half32,
                                           frag16& fa, frag16* fb)
{
  // A fragment 16x32 bf16 (ISA 7.12.2): this lane holds
  //   K in [8*half32, 8*half32+7] and [16+8*half32, 16+8*half32+7]
  fa.h[0] = *(const v8bf*)(arow + k0 + (half32 << 3));
  fa.h[1] = *(const v8bf*)(arow + k0 + 16 + (half32 << 3));
  #pragma unroll
  for (int j = 0; j < GEMM_NT; ++j) {
    // B fragment 32x16 bf16: this lane holds K [8*half32 .. +7] and [16+8*half32 .. +7]
    fb[j].h[0] = *(const v8bf*)(wrow[j] + k0 + (half32 << 3));
    fb[j].h[1] = *(const v8bf*)(wrow[j] + k0 + 16 + (half32 << 3));
  }
}

__device__ __forceinline__ void do_mmas(v8f* acc, const frag16& fa, const frag16* fb)
{
  #pragma unroll
  for (int j = 0; j < GEMM_NT; ++j) {
    acc[j] = __builtin_amdgcn_wmma_f32_16x16x32_bf16(false, fa.v, false, fb[j].v,
                                                     (short)0, acc[j], false, false);
  }
}

// ---------------------------------------------------------------------------
// GEMM on pre-converted bf16 operands:
//   A (M,K) bf16 row-major, W (N,K) bf16 row-major (i.e. h @ W.T), f32 accum.
// Requirements: K % 32 == 0, lda/ldw % 8 == 0 (b128-aligned rows), M,N >= 1.
// Load side is predicate-free (row/col clamped); bounds only at the store.
// Each wave computes a 16x64 strip. The K-loop is manually double-buffered:
// the next K-step's 10 b128 loads are issued before the current step's 4
// WMMAs, so WMMAs need only a partial loadcnt wait (latency overlapped).
// C[m][n] = act( (addsrc? addsrc[m,n]:0) + (bias? bias[n]:0) + sum_k A*W )
// ---------------------------------------------------------------------------
__global__ void wmma_gemm_bf16(const __bf16* __restrict__ A, const __bf16* __restrict__ W,
                               const float* __restrict__ bias, const float* __restrict__ addsrc,
                               float* __restrict__ C,
                               int M, int N, int K, int lda, int ldw, int ldc,
                               int mtiles, int ngroups, int act)
{
  const int wave = threadIdx.x >> 5;
  const int lane = threadIdx.x & 31;
  const int flat = blockIdx.x * (blockDim.x >> 5) + wave;
  if (flat >= mtiles * ngroups) return;
  const int mt = flat / ngroups;
  const int ng = flat - mt * ngroups;

  const int half32 = lane >> 4;       // 0/1 (wave32 lane halves)
  const int lr     = lane & 15;
  const int row0 = mt * 16;
  const int col0 = ng * (16 * GEMM_NT);

  // clamped operand rows -> unconditional aligned b128 loads
  const int am = min(row0 + lr, M - 1);
  const __bf16* __restrict__ arow = A + (size_t)am * lda;
  const __bf16* __restrict__ wrow[GEMM_NT];
  #pragma unroll
  for (int j = 0; j < GEMM_NT; ++j) {
    int bn = min(col0 + j * 16 + lr, N - 1);
    wrow[j] = W + (size_t)bn * ldw;
  }

  v8f acc[GEMM_NT] = {};
  frag16 fa0, fa1, fb0[GEMM_NT], fb1[GEMM_NT];

  load_frags(arow, wrow, 0, half32, fa0, fb0);
  int k0 = 0;
  for (;;) {
    int kn = k0 + 32;
    bool more = kn < K;
    if (more) load_frags(arow, wrow, kn, half32, fa1, fb1);   // prefetch next set
    do_mmas(acc, fa0, fb0);                                   // consume current set
    if (!more) break;
    k0 = kn; kn = k0 + 32; more = kn < K;
    if (more) load_frags(arow, wrow, kn, half32, fa0, fb0);
    do_mmas(acc, fa1, fb1);
    if (!more) break;
    k0 = kn;
  }

  // epilogue: store-side bounds only
  #pragma unroll
  for (int j = 0; j < GEMM_NT; ++j) {
    int ccol = col0 + j * 16 + lr;
    if (ccol < N) {
      float bv = bias ? bias[ccol] : 0.0f;
      #pragma unroll
      for (int r = 0; r < 8; ++r) {
        int crow = row0 + r + (half32 << 3);   // C/D: vgpr r -> row r + 8*(lane/16)
        if (crow < M) {
          float v = acc[j][r] + bv;
          if (addsrc) v += addsrc[(size_t)crow * ldc + ccol];
          if (act == 1) v = (v > 20.0f) ? v : log1pf(__expf(v));
          C[(size_t)crow * ldc + ccol] = v;
        }
      }
    }
  }
}

// ---------------------------------------------------------------------------
// generic f32 -> bf16 conversion (weights, once per launch)
// ---------------------------------------------------------------------------
__global__ void cvt_f32_bf16(const float* __restrict__ in, __bf16* __restrict__ out, int n)
{
  int idx = blockIdx.x * blockDim.x + threadIdx.x;
  if (idx < n) out[idx] = (__bf16)in[idx];
}

// ---------------------------------------------------------------------------
// pack all layers' W_dt (DEPTH,384,12) into zero-padded bf16 (DEPTH,384,32)
// ---------------------------------------------------------------------------
__global__ void pack_wdt_all(const float* __restrict__ wdt, __bf16* __restrict__ out)
{
  int idx = blockIdx.x * blockDim.x + threadIdx.x;
  if (idx >= DEPTH * D_INNER * KPAD) return;
  int k = idx & 31;
  int nl = idx >> 5;                       // l*384 + n
  out[idx] = (k < DT_RANK) ? (__bf16)wdt[nl * DT_RANK + k] : (__bf16)0.0f;
}

// ---------------------------------------------------------------------------
// pack dlt = xdbl[:, 0:12] into zero-padded bf16 (LTOK, 32)
// ---------------------------------------------------------------------------
__global__ void pack_dlt(const float* __restrict__ xdbl, __bf16* __restrict__ out)
{
  int idx = blockIdx.x * blockDim.x + threadIdx.x;
  if (idx >= LTOK * KPAD) return;
  int m = idx >> 5, k = idx & 31;
  out[idx] = (k < DT_RANK) ? (__bf16)xdbl[(long)m * XDBL_N + k] : (__bf16)0.0f;
}

// ---------------------------------------------------------------------------
// im2col for patch embedding (bf16 out): A[tok][c*256+p*16+q] = x[0,c,t,h*16+p,w*16+q]
// ---------------------------------------------------------------------------
__global__ void im2col_patch(const float* __restrict__ x, __bf16* __restrict__ A)
{
  int idx = blockIdx.x * blockDim.x + threadIdx.x;
  const int total = NPATCH * 768;
  if (idx >= total) return;
  int tok = idx / 768, k = idx - tok * 768;
  int c = k >> 8, rem = k & 255, p = rem >> 4, q = rem & 15;
  int t = tok / (GH * GW), r = tok - t * (GH * GW), h = r / GW, w = r - h * GW;
  A[idx] = (__bf16)x[(((long)c * SEQ_T + t) * IMGSZ + (h * 16 + p)) * IMGSZ + (w * 16 + q)];
}

// ---------------------------------------------------------------------------
// tok[0,:] = cls + pos[0,:]; tok[1+i,:] = patchC[i,:] + pos[1+i,:]
// ---------------------------------------------------------------------------
__global__ void assemble_tokens(const float* __restrict__ patchC, const float* __restrict__ cls,
                                const float* __restrict__ pos, float* __restrict__ tok)
{
  int idx = blockIdx.x * blockDim.x + threadIdx.x;
  if (idx >= LTOK * D_MODEL) return;
  int row = idx / D_MODEL, e = idx - row * D_MODEL;
  float v = (row == 0) ? cls[e] : patchC[(long)(row - 1) * D_MODEL + e];
  tok[idx] = v + pos[idx];
}

// ---------------------------------------------------------------------------
// LayerNorm: one wave per row, shfl reduction. Writes f32 and/or bf16.
// ---------------------------------------------------------------------------
__global__ void layernorm_rows(const float* __restrict__ x, const float* __restrict__ g,
                               const float* __restrict__ b,
                               float* __restrict__ out_f, __bf16* __restrict__ out_b, int cols)
{
  const int row = blockIdx.x, lane = threadIdx.x;
  const float* xr = x + (long)row * cols;
  float s = 0.f, ss = 0.f;
  for (int c = lane; c < cols; c += 32) { float v = xr[c]; s += v; ss += v * v; }
  for (int m = 16; m; m >>= 1) { s += __shfl_xor(s, m, 32); ss += __shfl_xor(ss, m, 32); }
  float mean = s / (float)cols;
  float var  = ss / (float)cols - mean * mean;
  float inv  = rsqrtf(var + 1e-5f);
  for (int c = lane; c < cols; c += 32) {
    float v = (xr[c] - mean) * inv * g[c] + b[c];
    if (out_f) out_f[(long)row * cols + c] = v;
    if (out_b) out_b[(long)row * cols + c] = (__bf16)v;
  }
}

// ---------------------------------------------------------------------------
// Depthwise causal conv (taps=4) over u = xz[:, :D_INNER], + bias, SiLU.
// Writes f32 (for scan) and bf16 (GEMM operand).
// ---------------------------------------------------------------------------
__global__ void conv_silu(const float* __restrict__ xz, const float* __restrict__ Wc,
                          const float* __restrict__ bc,
                          float* __restrict__ ucf, __bf16* __restrict__ ucb, int Lh)
{
  int idx = blockIdx.x * blockDim.x + threadIdx.x;
  if (idx >= Lh * D_INNER) return;
  int t = idx / D_INNER, d = idx - t * D_INNER;
  float acc = bc[d];
  #pragma unroll
  for (int k = 0; k < D_CONV; ++k) {
    int ts = t - (D_CONV - 1) + k;
    if (ts >= 0) acc += xz[(long)ts * (2 * D_INNER) + d] * Wc[d * D_CONV + k];
  }
  float v = siluf_(acc);
  ucf[idx] = v;
  ucb[idx] = (__bf16)v;
}

// ---------------------------------------------------------------------------
// Selective scan. One (d,n) state element per lane; 16 lanes = one channel.
// y[t,d] = (sum_n state*C + u*Dp) * silu(z), written as bf16 (GEMM operand).
// ---------------------------------------------------------------------------
__global__ void ssm_scan(const float* __restrict__ delta, const float* __restrict__ uc,
                         const float* __restrict__ xdbl, const float* __restrict__ xz,
                         const float* __restrict__ A_log, const float* __restrict__ Dp,
                         __bf16* __restrict__ y, int Lh)
{
  const int grp = threadIdx.x >> 4;               // 0..15 channel within block
  const int n   = threadIdx.x & 15;               // state index
  const int d   = blockIdx.x * 16 + grp;
  const float Av = -__expf(A_log[d * D_STATE + n]);
  const float Dv = Dp[d];
  float state = 0.f;
  for (int t = 0; t < Lh; ++t) {
    float dlt = delta[(long)t * D_INNER + d];
    float uu  = uc[(long)t * D_INNER + d];
    float Bv  = xdbl[(long)t * XDBL_N + DT_RANK + n];
    float Cv  = xdbl[(long)t * XDBL_N + DT_RANK + D_STATE + n];
    state = __expf(dlt * Av) * state + dlt * Bv * uu;
    float p = state * Cv;
    p += __shfl_xor(p, 1, 32); p += __shfl_xor(p, 2, 32);
    p += __shfl_xor(p, 4, 32); p += __shfl_xor(p, 8, 32);
    if (n == 0) {
      float zz = xz[(long)t * (2 * D_INNER) + D_INNER + d];
      y[(long)t * D_INNER + d] = (__bf16)((p + uu * Dv) * siluf_(zz));
    }
  }
}

// ---------------------------------------------------------------------------
extern "C" void kernel_launch(void* const* d_in, const int* in_sizes, int n_in,
                              void* d_out, int out_size, void* d_ws, size_t ws_size,
                              hipStream_t stream)
{
  (void)in_sizes; (void)n_in; (void)out_size; (void)ws_size;
  const float* x       = (const float*)d_in[0];
  const float* patch_w = (const float*)d_in[1];
  const float* patch_b = (const float*)d_in[2];
  const float* cls     = (const float*)d_in[3];
  const float* pos     = (const float*)d_in[4];
  const float* ln_g    = (const float*)d_in[5];
  const float* ln_b    = (const float*)d_in[6];
  const float* W_in    = (const float*)d_in[7];
  const float* Wc      = (const float*)d_in[8];
  const float* bc      = (const float*)d_in[9];
  const float* Wx      = (const float*)d_in[10];
  const float* W_dt    = (const float*)d_in[11];
  const float* b_dt    = (const float*)d_in[12];
  const float* A_log   = (const float*)d_in[13];
  const float* Dp      = (const float*)d_in[14];
  const float* W_out   = (const float*)d_in[15];
  const float* fn_g    = (const float*)d_in[16];
  const float* fn_b    = (const float*)d_in[17];

  // ---- workspace carving (256-byte aligned chunks) ----
  char* base = (char*)d_ws;
  size_t off = 0;
  auto alloc = [&](size_t bytes) -> void* {
    void* p = base + off;
    off += (bytes + 255) & ~(size_t)255;
    return p;
  };
  // f32 buffers
  float* patC  = (float*)alloc((size_t)NPATCH * D_MODEL * 4);
  float* c0    = (float*)alloc((size_t)LTOK * D_MODEL * 4);
  float* c1    = (float*)alloc((size_t)LTOK * D_MODEL * 4);
  float* xzbuf = (float*)alloc((size_t)LTOK * 2 * D_INNER * 4);
  float* ucbuf = (float*)alloc((size_t)LTOK * D_INNER * 4);
  float* xdbl  = (float*)alloc((size_t)LTOK * XDBL_N * 4);
  float* dlt   = (float*)alloc((size_t)LTOK * D_INNER * 4);
  // bf16 activation operands
  __bf16* Acol_b  = (__bf16*)alloc((size_t)NPATCH * 768 * 2);
  __bf16* h_b     = (__bf16*)alloc((size_t)LTOK * D_MODEL * 2);
  __bf16* uc_b    = (__bf16*)alloc((size_t)LTOK * D_INNER * 2);
  __bf16* dltin_b = (__bf16*)alloc((size_t)LTOK * KPAD * 2);
  __bf16* y_b     = (__bf16*)alloc((size_t)LTOK * D_INNER * 2);
  // bf16 weights (converted once per launch)
  __bf16* pw_b   = (__bf16*)alloc((size_t)D_MODEL * 768 * 2);
  __bf16* Win_b  = (__bf16*)alloc((size_t)DEPTH * 2 * D_INNER * D_MODEL * 2);
  __bf16* Wx_b   = (__bf16*)alloc((size_t)DEPTH * XDBL_N * D_INNER * 2);
  __bf16* Wdt_b  = (__bf16*)alloc((size_t)DEPTH * D_INNER * KPAD * 2);
  __bf16* Wout_b = (__bf16*)alloc((size_t)DEPTH * D_MODEL * D_INNER * 2);

  auto cvt = [&](const float* in, __bf16* out, int n) {
    cvt_f32_bf16<<<(n + 255) / 256, 256, 0, stream>>>(in, out, n);
  };
  auto gemm = [&](const __bf16* A, const __bf16* W, const float* bias, const float* add,
                  float* C, int M, int N, int K, int lda, int ldw, int ldc, int act) {
    int mtiles  = (M + 15) / 16;
    int ngroups = (N + 16 * GEMM_NT - 1) / (16 * GEMM_NT);
    int waves   = mtiles * ngroups;
    int blocks  = (waves + 7) / 8;                       // 8 waves / 256-thread block
    wmma_gemm_bf16<<<blocks, 256, 0, stream>>>(A, W, bias, add, C, M, N, K,
                                               lda, ldw, ldc, mtiles, ngroups, act);
  };

  // ---- one-time (per launch) weight conversion ----
  cvt(patch_w, pw_b, D_MODEL * 768);
  cvt(W_in,  Win_b,  DEPTH * 2 * D_INNER * D_MODEL);
  cvt(Wx,    Wx_b,   DEPTH * XDBL_N * D_INNER);
  cvt(W_out, Wout_b, DEPTH * D_MODEL * D_INNER);
  {
    int tw = DEPTH * D_INNER * KPAD;
    pack_wdt_all<<<(tw + 255) / 256, 256, 0, stream>>>(W_dt, Wdt_b);
  }

  // ---- patch embedding ----
  {
    int tot = NPATCH * 768;
    im2col_patch<<<(tot + 255) / 256, 256, 0, stream>>>(x, Acol_b);
    gemm(Acol_b, pw_b, patch_b, nullptr, patC, NPATCH, D_MODEL, 768, 768, 768, D_MODEL, 0);
    int tt = LTOK * D_MODEL;
    assemble_tokens<<<(tt + 255) / 256, 256, 0, stream>>>(patC, cls, pos, c0);
  }

  // ---- Mamba blocks ----
  float* cur = c0; float* nxt = c1;
  for (int l = 0; l < DEPTH; ++l) {
    const float*  g   = ln_g  + (size_t)l * D_MODEL;
    const float*  bb  = ln_b  + (size_t)l * D_MODEL;
    const __bf16* Wi  = Win_b  + (size_t)l * 2 * D_INNER * D_MODEL;
    const float*  Wcl = Wc    + (size_t)l * D_INNER * D_CONV;
    const float*  bcl = bc    + (size_t)l * D_INNER;
    const __bf16* Wxl = Wx_b   + (size_t)l * XDBL_N * D_INNER;
    const __bf16* Wdl = Wdt_b  + (size_t)l * D_INNER * KPAD;
    const float*  bdt = b_dt  + (size_t)l * D_INNER;
    const float*  Al  = A_log + (size_t)l * D_INNER * D_STATE;
    const float*  Dl  = Dp    + (size_t)l * D_INNER;
    const __bf16* Wo  = Wout_b + (size_t)l * D_MODEL * D_INNER;

    layernorm_rows<<<LTOK, 32, 0, stream>>>(cur, g, bb, nullptr, h_b, D_MODEL);
    // xz = h @ W_in.T   (1569 x 768)
    gemm(h_b, Wi, nullptr, nullptr, xzbuf, LTOK, 2 * D_INNER, D_MODEL,
         D_MODEL, D_MODEL, 2 * D_INNER, 0);
    {
      int tot = LTOK * D_INNER;
      conv_silu<<<(tot + 255) / 256, 256, 0, stream>>>(xzbuf, Wcl, bcl, ucbuf, uc_b, LTOK);
    }
    // xdbl = u_conv @ Wx.T   (1569 x 44)
    gemm(uc_b, Wxl, nullptr, nullptr, xdbl, LTOK, XDBL_N, D_INNER,
         D_INNER, D_INNER, XDBL_N, 0);
    // pack K=12 activation operand to K=32 zero-padded bf16
    {
      int tot = LTOK * KPAD;
      pack_dlt<<<(tot + 255) / 256, 256, 0, stream>>>(xdbl, dltin_b);
    }
    // delta = softplus(dlt @ W_dt.T + b_dt)   (1569 x 384)
    gemm(dltin_b, Wdl, bdt, nullptr, dlt, LTOK, D_INNER, KPAD,
         KPAD, KPAD, D_INNER, 1);
    // sequential selective scan + gating (bf16 output = next GEMM operand)
    ssm_scan<<<D_INNER / 16, 256, 0, stream>>>(dlt, ucbuf, xdbl, xzbuf, Al, Dl, y_b, LTOK);
    // out = residual + y @ W_out.T
    gemm(y_b, Wo, nullptr, cur, nxt, LTOK, D_MODEL, D_INNER,
         D_INNER, D_INNER, D_MODEL, 0);
    float* tmp = cur; cur = nxt; nxt = tmp;
  }

  // ---- final LayerNorm into d_out ----
  layernorm_rows<<<LTOK, 32, 0, stream>>>(cur, fn_g, fn_b, (float*)d_out, nullptr, D_MODEL);
}
